// LundNet_55044300866329
// MI455X (gfx1250) — compile-verified, matrix operand order checked by reading, not run on hardware
//
#include <hip/hip_runtime.h>
#include <hip/hip_bf16.h>

#define NN 262144   // nodes
#define NE 524288   // edges
#define NG 2048     // graphs
#define CDIV(a,b) (((a)+(b)-1)/(b))

typedef __attribute__((ext_vector_type(16))) __bf16 v16bf;
typedef __attribute__((ext_vector_type(8)))  float  v8f;

// ---------------- device helpers ----------------

static __device__ __forceinline__ unsigned short f2b(float f) {
  unsigned int u = __float_as_uint(f);
  u += 0x7FFFu + ((u >> 16) & 1u);           // round-to-nearest-even
  return (unsigned short)(u >> 16);
}

// Load one bf16 WMMA fragment slice for this lane. p points at element
// [row_or_col][k0 + half*8]; j=0..7 <- K=half*8+j, j=8..15 <- K=16+half*8+(j-8)
// (CDNA5 16-bit A/B VGPR layout).
static __device__ __forceinline__ v16bf load_frag(const unsigned short* p) {
  union { uint4 u[2]; v16bf f; } r;
  r.u[0] = *reinterpret_cast<const uint4*>(p);
  r.u[1] = *reinterpret_cast<const uint4*>(p + 16);
  return r.f;
}

static __device__ __forceinline__ v8f wmma_bf16(v16bf a, v16bf b, v8f c) {
  return __builtin_amdgcn_wmma_f32_16x16x32_bf16(false, a, false, b, (short)0, c,
                                                 false, false);
}

// ---------------- generic WMMA GEMM ----------------
// Y[M, Nc] = epilogue( A[M, lda-strided, K] * B[Nc, K]^T )
// epilogue: (+bias[col]) -> (relu if mode) -> (*scale[col] + shift[col]) ->
//           store f32 / bf16, or atomic segment-sum into gsum[gid[row]*ldc+col].

struct EpiArgs {
  const float* bias;
  const float* scale;
  const float* shift;
  float* Cf;
  unsigned short* Cb;
  int ldc;
  const int* gid;
  float* gsum;
  int mode;   // 0 = linear, 1 = relu
};

static __device__ __forceinline__ void store_tile(const v8f& acc, long mBase, int nBase,
                                                  int half, int idx, const EpiArgs& e) {
  const int col = nBase + idx;
  const float bv = e.bias  ? e.bias[col]  : 0.f;
  float sv = 1.f, tv = 0.f;
  if (e.scale) { sv = e.scale[col]; tv = e.shift[col]; }
#pragma unroll
  for (int r = 0; r < 8; ++r) {
    const long row = mBase + (half << 3) + r;   // C/D layout: VGPR r -> M=r+half*8
    float y = acc[r] + bv;
    if (e.mode) y = fmaxf(y, 0.f);
    if (e.scale) y = y * sv + tv;
    if (e.gsum) {
      unsafeAtomicAdd(&e.gsum[(long)e.gid[row] * e.ldc + col], y);
    } else {
      const long o = row * (long)e.ldc + col;
      if (e.Cf) e.Cf[o] = y;
      if (e.Cb) e.Cb[o] = f2b(y);
    }
  }
}

// Block tile 128(M) x 64(N); 8 waves in a 4(M) x 2(N) grid, 32x32 per wave.
// B chunks (64 cols x 32 k) staged through LDS, double-buffered, rows padded
// to 40 elements (80 B) for conflict-free ds_load_b128 fragment reads.
// Out-of-range B rows are loaded from a clamped (valid) address and zeroed
// with cndmask -- keeps the K-loop completely branch/EXEC-mask free.
#define BROW 40

__global__ __launch_bounds__(256) void gemm_wmma(
    const unsigned short* __restrict__ A, int lda,
    const unsigned short* __restrict__ B, int K,
    int Nc, EpiArgs e) {
  __shared__ unsigned short Bs[2][64 * BROW];

  const int tid  = threadIdx.x;
  const int wave = __builtin_amdgcn_readfirstlane(tid) >> 5;  // SGPR wave id
  const int lane = tid & 31;
  const int half = lane >> 4, idx = lane & 15;
  const long m0 = (long)blockIdx.y * 128 + (long)(wave >> 1) * 32;
  const int  nBlock = blockIdx.x * 64;
  const int  n0 = nBlock + (wave & 1) * 32;
  const bool v0 = n0 < Nc;                  // scalar (wave-uniform, SGPR)
  const bool v1 = (n0 + 16) < Nc;

  // cooperative B staging: 4 threads per 32-element row, 16 B each
  const int brow = tid >> 2;
  const int bco  = (tid & 3) << 3;
  const bool bvalid = (nBlock + brow) < Nc;
  const int browC = bvalid ? (nBlock + brow) : (Nc - 1);   // clamped, always valid
  const unsigned short* gB = B + (long)browC * K + bco;
  unsigned short* sB = &Bs[0][brow * BROW + bco];

  v8f acc00 = {}, acc01 = {}, acc10 = {}, acc11 = {};
  const unsigned short* pa0 = A + (m0 + idx) * (long)lda + (half << 3);
  const unsigned short* pa1 = pa0 + 16 * (long)lda;
  const unsigned short* lb0 = &Bs[0][((wave & 1) * 32 + idx) * BROW + (half << 3)];

  {   // preload chunk 0
    uint4 v = *reinterpret_cast<const uint4*>(gB);
    v.x = bvalid ? v.x : 0u;  v.y = bvalid ? v.y : 0u;
    v.z = bvalid ? v.z : 0u;  v.w = bvalid ? v.w : 0u;
    *reinterpret_cast<uint4*>(sB) = v;
  }
  __syncthreads();

  for (int k0 = 0; k0 < K; k0 += 32) {
    const int buf = (k0 >> 5) & 1;
    const unsigned short* bp = lb0 + buf * (64 * BROW);
    v16bf a0 = load_frag(pa0 + k0);
    v16bf a1 = load_frag(pa1 + k0);
    v16bf b0 = load_frag(bp);
    v16bf b1 = load_frag(bp + 16 * BROW);
    if (k0 + 32 < K) {                      // prefetch next chunk -> other buffer
      uint4 v = *reinterpret_cast<const uint4*>(gB + k0 + 32);
      v.x = bvalid ? v.x : 0u;  v.y = bvalid ? v.y : 0u;
      v.z = bvalid ? v.z : 0u;  v.w = bvalid ? v.w : 0u;
      *reinterpret_cast<uint4*>(sB + (buf ^ 1) * (64 * BROW)) = v;
    }
    acc00 = wmma_bf16(a0, b0, acc00);
    acc10 = wmma_bf16(a1, b0, acc10);
    acc01 = wmma_bf16(a0, b1, acc01);       // zero columns if n-range invalid
    acc11 = wmma_bf16(a1, b1, acc11);
    __syncthreads();
  }

  if (v0) {
    store_tile(acc00, m0,      n0, half, idx, e);
    store_tile(acc10, m0 + 16, n0, half, idx, e);
    if (v1) {
      store_tile(acc01, m0,      n0 + 16, half, idx, e);
      store_tile(acc11, m0 + 16, n0 + 16, half, idx, e);
    }
  }
}

// ---------------- support kernels ----------------

__global__ void k_deg(const int* __restrict__ dst, float* __restrict__ deg) {
  int e = blockIdx.x * 256 + threadIdx.x;
  if (e < NE) unsafeAtomicAdd(&deg[dst[e]], 1.0f);
}

__global__ void k_bn_input(const float* __restrict__ x, const float* __restrict__ beta,
                           const float* __restrict__ gamma, const float* __restrict__ mean,
                           const float* __restrict__ var, unsigned short* __restrict__ out) {
  int n = blockIdx.x * 256 + threadIdx.x;
  if (n >= NN) return;
#pragma unroll
  for (int j = 0; j < 32; ++j) {
    float y = 0.f;
    if (j < 5)
      y = (x[n * 5 + j] - mean[j]) * (gamma[j] * rsqrtf(var[j] + 1e-5f)) + beta[j];
    out[n * 32 + j] = f2b(y);                 // zero-pad K=5 -> 32
  }
}

// Build stacked conv weights: rows [0,c): s*theta ; rows [c,2c): s*(phi-theta).
// biasC = (b_theta + b_phi)*s + beta - mean*s  (BN folded).
__global__ void k_prep_conv(const float* __restrict__ tw, const float* __restrict__ tb,
                            const float* __restrict__ pw, const float* __restrict__ pb,
                            const float* __restrict__ beta, const float* __restrict__ gamma,
                            const float* __restrict__ mean, const float* __restrict__ var,
                            int cin, int kp, int c,
                            unsigned short* __restrict__ W, float* __restrict__ biasC) {
  int t = blockIdx.x * 256 + threadIdx.x;
  if (t >= 2 * c * kp) return;
  int r = t / kp, k = t % kp;
  int rr = (r < c) ? r : r - c;
  float s = gamma[rr] * rsqrtf(var[rr] + 1e-5f);
  float w = 0.f;
  if (k < cin) {
    float th = tw[rr * cin + k];
    w = (r < c) ? s * th : s * (pw[rr * cin + k] - th);
  }
  W[t] = f2b(w);
  if (k == 0 && r < c) biasC[r] = (tb[r] + pb[r]) * s + beta[r] - mean[r] * s;
}

// Shortcut weights (no bias in linear): W' = s*W, shift = beta - mean*s.
__global__ void k_prep_sc(const float* __restrict__ w0, const float* __restrict__ beta,
                          const float* __restrict__ gamma, const float* __restrict__ mean,
                          const float* __restrict__ var, int cin, int kp, int c,
                          unsigned short* __restrict__ W, float* __restrict__ shift) {
  int t = blockIdx.x * 256 + threadIdx.x;
  if (t >= c * kp) return;
  int r = t / kp, k = t % kp;
  float s = gamma[r] * rsqrtf(var[r] + 1e-5f);
  W[t] = f2b(k < cin ? s * w0[r * cin + k] : 0.f);
  if (k == 0) shift[r] = beta[r] - mean[r] * s;
}

__global__ void k_prep_plain(const float* __restrict__ w, int n, unsigned short* __restrict__ W) {
  int t = blockIdx.x * 256 + threadIdx.x;
  if (t < n) W[t] = f2b(w[t]);
}

__global__ void k_prep_bnvec(const float* __restrict__ beta, const float* __restrict__ gamma,
                             const float* __restrict__ mean, const float* __restrict__ var,
                             int c, float* __restrict__ s, float* __restrict__ t0) {
  int i = blockIdx.x * 256 + threadIdx.x;
  if (i >= c) return;
  float sv = gamma[i] * rsqrtf(var[i] + 1e-5f);
  s[i] = sv;
  t0[i] = beta[i] - mean[i] * sv;
}

// scatter agg[dst] += Q[src]  (Q = second half of TQ), 4 channels per thread
__global__ void k_scatter(const int* __restrict__ src, const int* __restrict__ dst,
                          const float* __restrict__ TQ, int twoC, int c, int chunks,
                          float* __restrict__ agg) {
  long t = (long)blockIdx.x * 256 + threadIdx.x;
  if (t >= (long)NE * chunks) return;
  int e = (int)(t / chunks);
  int ch = (int)(t % chunks) << 2;
  int s = src[e], d = dst[e];
  const float4 q = *reinterpret_cast<const float4*>(TQ + (long)s * twoC + c + ch);
  float* b = agg + (long)d * c + ch;
  unsafeAtomicAdd(b + 0, q.x);
  unsafeAtomicAdd(b + 1, q.y);
  unsafeAtomicAdd(b + 2, q.z);
  unsafeAtomicAdd(b + 3, q.w);
}

// h = relu(T' + agg/deg + bias'), 0 if deg==0 ; write f32 + bf16
__global__ void k_fin_conv(const float* __restrict__ TQ, int twoC,
                           const float* __restrict__ agg, const float* __restrict__ deg,
                           const float* __restrict__ biasC, int c,
                           float* __restrict__ hf, unsigned short* __restrict__ hb) {
  long t = (long)blockIdx.x * 256 + threadIdx.x;
  if (t >= (long)NN * c) return;
  int n = (int)(t / c), j = (int)(t % c);
  float dg = deg[n];
  float y = 0.f;
  if (dg > 0.5f)
    y = fmaxf(TQ[(long)n * twoC + j] + agg[t] / dg + biasC[j], 0.f);
  hf[t] = y;
  hb[t] = f2b(y);
}

// block output: relu(h2 + sc) written into concat slices (f32 + bf16)
__global__ void k_fin_block(const float* __restrict__ TQ, int twoC,
                            const float* __restrict__ agg, const float* __restrict__ deg,
                            const float* __restrict__ biasC, int c,
                            const float* __restrict__ scSrc, int scLd,
                            float* __restrict__ outF, unsigned short* __restrict__ outB) {
  long t = (long)blockIdx.x * 256 + threadIdx.x;
  if (t >= (long)NN * c) return;
  int n = (int)(t / c), j = (int)(t % c);
  float dg = deg[n];
  float h = 0.f;
  if (dg > 0.5f)
    h = fmaxf(TQ[(long)n * twoC + j] + agg[(long)n * c + j] / dg + biasC[j], 0.f);
  float y = fmaxf(h + scSrc[(long)n * scLd + j], 0.f);
  outF[(long)n * 448 + j] = y;
  outB[(long)n * 448 + j] = f2b(y);
}

__global__ void k_count(const int* __restrict__ gid, float* __restrict__ cnt) {
  int n = blockIdx.x * 256 + threadIdx.x;
  if (n < NN) unsafeAtomicAdd(&cnt[gid[n]], 1.0f);
}

__global__ void k_mean(const float* __restrict__ gsum, const float* __restrict__ cnt,
                       unsigned short* __restrict__ gbf) {
  int t = blockIdx.x * 256 + threadIdx.x;
  if (t >= NG * 384) return;
  float g = gsum[t] / fmaxf(cnt[t / 384], 1.0f);
  gbf[t] = f2b(g);
}

__global__ void k_fc2(const float* __restrict__ h, const float* __restrict__ w,
                      const float* __restrict__ b, float* __restrict__ out) {
  int t = blockIdx.x * 256 + threadIdx.x;
  if (t >= NG * 2) return;
  int m = t >> 1, o = t & 1;
  float s = b[o];
  for (int k = 0; k < 128; ++k) s += h[m * 128 + k] * w[o * 128 + k];
  out[t] = s;
}

// ---------------- host ----------------

static void launch_gemm(hipStream_t st, const unsigned short* A, int lda,
                        const unsigned short* B, int K, int M, int Nc, EpiArgs e) {
  dim3 g(CDIV(Nc, 64), M / 128);
  gemm_wmma<<<g, dim3(256), 0, st>>>(A, lda, B, K, Nc, e);
}

extern "C" void kernel_launch(void* const* d_in, const int* in_sizes, int n_in,
                              void* d_out, int out_size, void* d_ws, size_t ws_size,
                              hipStream_t stream) {
  (void)in_sizes; (void)n_in; (void)out_size; (void)ws_size;
  const float* features = (const float*)d_in[0];
  const int* src  = (const int*)d_in[1];
  const int* dst  = (const int*)d_in[2];
  const int* gids = (const int*)d_in[3];
  auto FP = [&](int i) { return (const float*)d_in[i]; };

  // ---- parameter leaf indices (jax tree_leaves: dict keys sorted) ----
  struct Conv { int cin, cout, kp;
                int iBeta, iGamma, iMean, iVar, iPhiB, iPhiW, iThetaB, iThetaW; };
  const int chans[6][2] = {{32,32},{32,32},{64,64},{64,64},{128,128},{128,128}};
  Conv cv[12];
  int scWi[6] = {0}, scBi[6] = {0}; bool hasSc[6];
  int inFeat[6], idx = 5, in_f = 5;
  for (int b = 0; b < 6; ++b) {
    inFeat[b] = in_f;
    for (int j = 0; j < 2; ++j) {
      Conv c; c.cin = (j == 0 ? in_f : chans[b][0]); c.cout = chans[b][j];
      c.kp = CDIV(c.cin, 32) * 32;
      c.iBeta = idx; c.iGamma = idx + 1; c.iMean = idx + 2; c.iVar = idx + 3;
      c.iPhiB = idx + 4; c.iPhiW = idx + 5; c.iThetaB = idx + 6; c.iThetaW = idx + 7;
      idx += 8;
      cv[2 * b + j] = c;
    }
    hasSc[b] = (in_f != chans[b][1]);
    if (hasSc[b]) { scWi[b] = idx; scBi[b] = idx + 1; idx += 5; }
    in_f = chans[b][1];
  }
  const int iFtsBeta = idx;       idx += 4;
  const int iFc1B = idx, iFc1W = idx + 1; idx += 2;
  const int iFc2B = idx, iFc2W = idx + 1; idx += 2;
  const int iFusB = idx, iFusBeta = idx + 1, iFusW = idx + 5;

  // ---- workspace carve-up ----
  size_t off = 0;
  auto arena = [&](size_t bytes) -> void* {
    void* p = (char*)d_ws + off;
    off += (bytes + 255) & ~(size_t)255;
    return p;
  };
  float*          deg     = (float*)arena((size_t)NN * 4);
  unsigned short* fts0    = (unsigned short*)arena((size_t)NN * 32 * 2);
  float*          concatF = (float*)arena((size_t)NN * 448 * 4);
  unsigned short* concatB = (unsigned short*)arena((size_t)NN * 448 * 2);
  float*          TQ      = (float*)arena((size_t)NN * 256 * 4);
  float*          agg     = (float*)arena((size_t)NN * 128 * 4);
  float*          h1f     = (float*)arena((size_t)NN * 128 * 4);
  unsigned short* h1b     = (unsigned short*)arena((size_t)NN * 128 * 2);
  float*          scf     = (float*)arena((size_t)NN * 128 * 4);
  float*          gsum    = (float*)arena((size_t)NG * 384 * 4);
  float*          gcnt    = (float*)arena((size_t)NG * 4);
  unsigned short* gbf     = (unsigned short*)arena((size_t)NG * 384 * 2);
  float*          fc1f    = (float*)arena((size_t)NG * 128 * 4);
  unsigned short* Wstk[12]; float* biasC[12];
  for (int k = 0; k < 12; ++k) {
    Wstk[k]  = (unsigned short*)arena((size_t)2 * cv[k].cout * cv[k].kp * 2);
    biasC[k] = (float*)arena((size_t)cv[k].cout * 4);
  }
  unsigned short* Wsc[6]; float* scT[6];
  for (int b = 0; b < 6; ++b) {
    if (!hasSc[b]) { Wsc[b] = nullptr; scT[b] = nullptr; continue; }
    int kp = CDIV(inFeat[b], 32) * 32;
    Wsc[b] = (unsigned short*)arena((size_t)chans[b][1] * kp * 2);
    scT[b] = (float*)arena((size_t)chans[b][1] * 4);
  }
  unsigned short* Wfus = (unsigned short*)arena((size_t)384 * 448 * 2);
  unsigned short* Wfc1 = (unsigned short*)arena((size_t)128 * 384 * 2);
  float* fusS = (float*)arena(384 * 4);
  float* fusT = (float*)arena(384 * 4);

  // ---- weight prep (BN folded, f32 -> bf16) ----
  for (int k = 0; k < 12; ++k) {
    const Conv& c = cv[k];
    int tot = 2 * c.cout * c.kp;
    k_prep_conv<<<CDIV(tot, 256), 256, 0, stream>>>(
        FP(c.iThetaW), FP(c.iThetaB), FP(c.iPhiW), FP(c.iPhiB),
        FP(c.iBeta), FP(c.iGamma), FP(c.iMean), FP(c.iVar),
        c.cin, c.kp, c.cout, Wstk[k], biasC[k]);
  }
  for (int b = 0; b < 6; ++b) {
    if (!hasSc[b]) continue;
    int kp = CDIV(inFeat[b], 32) * 32, c = chans[b][1];
    k_prep_sc<<<CDIV(c * kp, 256), 256, 0, stream>>>(
        FP(scWi[b]), FP(scBi[b]), FP(scBi[b] + 1), FP(scBi[b] + 2), FP(scBi[b] + 3),
        inFeat[b], kp, c, Wsc[b], scT[b]);
  }
  k_prep_plain<<<CDIV(384 * 448, 256), 256, 0, stream>>>(FP(iFusW), 384 * 448, Wfus);
  k_prep_plain<<<CDIV(128 * 384, 256), 256, 0, stream>>>(FP(iFc1W), 128 * 384, Wfc1);
  k_prep_bnvec<<<CDIV(384, 256), 256, 0, stream>>>(
      FP(iFusBeta), FP(iFusBeta + 1), FP(iFusBeta + 2), FP(iFusBeta + 3), 384, fusS, fusT);

  // ---- degrees + input BN ----
  hipMemsetAsync(deg, 0, (size_t)NN * 4, stream);
  k_deg<<<CDIV(NE, 256), 256, 0, stream>>>(dst, deg);
  k_bn_input<<<CDIV(NN, 256), 256, 0, stream>>>(
      features, FP(iFtsBeta), FP(iFtsBeta + 1), FP(iFtsBeta + 2), FP(iFtsBeta + 3), fts0);

  // ---- conv blocks ----
  int colOff[6]; { int o = 0; for (int b = 0; b < 6; ++b) { colOff[b] = o; o += chans[b][1]; } }
  for (int b = 0; b < 6; ++b) {
    const Conv& c1 = cv[2 * b];
    const Conv& c2 = cv[2 * b + 1];
    const unsigned short* Ain = (b == 0) ? fts0 : concatB + colOff[b - 1];
    const int ldin = (b == 0) ? 32 : 448;

    // conv1: TQ = [T' | Q] node projections
    EpiArgs e1{}; e1.Cf = TQ; e1.ldc = 2 * c1.cout;
    launch_gemm(stream, Ain, ldin, Wstk[2 * b], c1.kp, NN, 2 * c1.cout, e1);
    hipMemsetAsync(agg, 0, (size_t)NN * c1.cout * 4, stream);
    k_scatter<<<CDIV(NE * (c1.cout / 4), 256), 256, 0, stream>>>(
        src, dst, TQ, 2 * c1.cout, c1.cout, c1.cout / 4, agg);
    k_fin_conv<<<CDIV(NN * c1.cout, 256), 256, 0, stream>>>(
        TQ, 2 * c1.cout, agg, deg, biasC[2 * b], c1.cout, h1f, h1b);

    // conv2
    EpiArgs e2{}; e2.Cf = TQ; e2.ldc = 2 * c2.cout;
    launch_gemm(stream, h1b, c1.cout, Wstk[2 * b + 1], c2.kp, NN, 2 * c2.cout, e2);
    hipMemsetAsync(agg, 0, (size_t)NN * c2.cout * 4, stream);
    k_scatter<<<CDIV(NE * (c2.cout / 4), 256), 256, 0, stream>>>(
        src, dst, TQ, 2 * c2.cout, c2.cout, c2.cout / 4, agg);

    // shortcut
    const float* scSrc; int scLd;
    if (hasSc[b]) {
      EpiArgs es{}; es.bias = scT[b]; es.Cf = scf; es.ldc = c2.cout;
      int kp = CDIV(inFeat[b], 32) * 32;
      launch_gemm(stream, Ain, ldin, Wsc[b], kp, NN, c2.cout, es);
      scSrc = scf; scLd = c2.cout;
    } else {
      scSrc = concatF + colOff[b - 1]; scLd = 448;
    }
    k_fin_block<<<CDIV(NN * c2.cout, 256), 256, 0, stream>>>(
        TQ, 2 * c2.cout, agg, deg, biasC[2 * b + 1], c2.cout,
        scSrc, scLd, concatF + colOff[b], concatB + colOff[b]);
  }

  // ---- fusion (relu -> BN) fused with per-graph segment-sum ----
  hipMemsetAsync(gsum, 0, (size_t)NG * 384 * 4, stream);
  hipMemsetAsync(gcnt, 0, (size_t)NG * 4, stream);
  k_count<<<CDIV(NN, 256), 256, 0, stream>>>(gids, gcnt);
  EpiArgs ef{}; ef.bias = FP(iFusB); ef.scale = fusS; ef.shift = fusT; ef.mode = 1;
  ef.gid = gids; ef.gsum = gsum; ef.ldc = 384;
  launch_gemm(stream, concatB, 448, Wfus, 448, NN, 384, ef);
  k_mean<<<CDIV(NG * 384, 256), 256, 0, stream>>>(gsum, gcnt, gbf);

  // ---- classifier head ----
  EpiArgs e5{}; e5.bias = FP(iFc1B); e5.mode = 1; e5.Cf = fc1f; e5.ldc = 128;
  launch_gemm(stream, gbf, 384, Wfc1, 384, NG, 128, e5);
  k_fc2<<<CDIV(NG * 2, 256), 256, 0, stream>>>(fc1f, FP(iFc2W), FP(iFc2B), (float*)d_out);
}
